// DGCNN_51831665328279
// MI455X (gfx1250) — matrix operand order repaired and checked by reading, not compiled, not run on hardware
//
#include <hip/hip_runtime.h>
#include <hip/hip_bf16.h>
#include <stdint.h>

#define KNN_K 20
#define NPTS  2048
#define BATCH 8
#define BNEPS 1e-5f

typedef __attribute__((ext_vector_type(16))) _Float16 v16h;
typedef __attribute__((ext_vector_type(8)))  float    v8f;
// 16B-aligned view for LDS vector loads (row stride is 16B- but not 32B-aligned)
typedef v16h v16h_a16 __attribute__((aligned(16)));

// ---------- helpers ----------
__device__ __forceinline__ unsigned encodeF(float f) {
    unsigned u = __float_as_uint(f);
    return (u & 0x80000000u) ? ~u : (u | 0x80000000u);   // monotone float->uint
}
__device__ __forceinline__ float decodeF(unsigned u) {
    u = (u & 0x80000000u) ? (u & 0x7fffffffu) : ~u;
    return __uint_as_float(u);
}
__device__ __forceinline__ float lrelu(float y) { return y > 0.f ? y : 0.2f * y; }

// ---------- pack fp32 weights into per-lane WMMA A-fragment order (fp16) ----------
// Layout: Wp[((mt*KT + cb)*32 + lane)*16 + j], so each lane's 16 halves are one
// contiguous 32B v16h; K = cb*32 + hi*16 + j
__global__ void packW_kernel(const float* __restrict__ W, _Float16* __restrict__ Wp,
                             int O, int C2) {
    int i = blockIdx.x * blockDim.x + threadIdx.x;
    if (i >= O * C2) return;
    int j    = i & 15;
    int lane = (i >> 4) & 31;
    int rest = i >> 9;              // mt*KT + cb
    int KT   = C2 >> 5;
    int cb   = rest % KT, mt = rest / KT;
    int row = lane & 15, hi = lane >> 4;
    int vv = j >> 1, p = j & 1;
    int k = cb * 32 + hi * 16 + j; (void)vv; (void)p;
    Wp[i] = (_Float16)W[(size_t)(mt * 16 + row) * C2 + k];
}

// ---------- per-point sum of squares (for KNN distances) ----------
__global__ void sumsq_kernel(const float* __restrict__ x, float* __restrict__ xx, int C) {
    int i = blockIdx.x * blockDim.x + threadIdx.x;
    if (i >= BATCH * NPTS) return;
    int b = i / NPTS, n = i - b * NPTS;
    const float* xp = x + (size_t)b * C * NPTS + n;
    float s = 0.f;
    for (int c = 0; c < C; ++c) { float v = xp[c * NPTS]; s += v * v; }
    xx[i] = s;
}

// ---------- KNN: one wave32 per point; per-lane top-20 + wave merge ----------
__global__ __launch_bounds__(128) void knn_kernel(const float* __restrict__ x,
                                                  const float* __restrict__ xx,
                                                  int* __restrict__ idxOut, int C) {
    __shared__ float xns[4][128];
    __shared__ float sv[4][32][KNN_K];
    __shared__ int   si[4][32][KNN_K];
    int b = blockIdx.y;
    int w = threadIdx.x >> 5, lane = threadIdx.x & 31;
    int n = blockIdx.x * 4 + w;
    const float* xb = x + (size_t)b * C * NPTS;
    for (int c = lane; c < C; c += 32) xns[w][c] = xb[c * NPTS + n];
    __syncthreads();
    float xxn = xx[b * NPTS + n];

    float v[KNN_K]; int id[KNN_K];
    #pragma unroll
    for (int t = 0; t < KNN_K; ++t) { v[t] = -3.4e38f; id[t] = 0; }

    for (int m = lane; m < NPTS; m += 32) {
        float dot = 0.f;
        for (int c = 0; c < C; ++c) dot = fmaf(xns[w][c], xb[c * NPTS + m], dot);
        float pd = 2.f * dot - xxn - xx[b * NPTS + m];   // = -||xn-xm||^2
        if (pd > v[KNN_K - 1]) {
            v[KNN_K - 1] = pd; id[KNN_K - 1] = m;
            #pragma unroll
            for (int t = KNN_K - 1; t > 0; --t) {
                if (v[t] > v[t - 1]) {
                    float tv = v[t]; v[t] = v[t - 1]; v[t - 1] = tv;
                    int ti = id[t]; id[t] = id[t - 1]; id[t - 1] = ti;
                }
            }
        }
    }
    #pragma unroll
    for (int t = 0; t < KNN_K; ++t) { sv[w][lane][t] = v[t]; si[w][lane][t] = id[t]; }
    __syncthreads();

    int ptr = 0;
    for (int it = 0; it < KNN_K; ++it) {
        float cv = sv[w][lane][ptr];
        int   ci = si[w][lane][ptr];
        int   cl = lane;
        #pragma unroll
        for (int off = 16; off > 0; off >>= 1) {
            float ov = __shfl_xor(cv, off, 32);
            int   oi = __shfl_xor(ci, off, 32);
            int   ol = __shfl_xor(cl, off, 32);
            if (ov > cv || (ov == cv && ol < cl)) { cv = ov; ci = oi; cl = ol; }
        }
        if (lane == cl) ptr++;
        if (lane == 0) idxOut[((size_t)b * NPTS + n) * KNN_K + it] = ci;
    }
}

// ---------- layer 1 (C2=6 too small for WMMA): scalar fused edge conv ----------
__global__ __launch_bounds__(64) void edge1_kernel(const float* __restrict__ x,
                                                   const int* __restrict__ idx,
                                                   const float* __restrict__ W1,
                                                   unsigned* __restrict__ hmaxEnc,
                                                   float* __restrict__ chanSum,
                                                   float* __restrict__ chanSumSq) {
    __shared__ float ctr[3];
    __shared__ float nbr[KNN_K][3];
    int n = blockIdx.x, b = blockIdx.y, tid = threadIdx.x;
    const float* xb = x + (size_t)b * 3 * NPTS;
    if (tid < 3) ctr[tid] = xb[tid * NPTS + n];
    if (tid < KNN_K * 3) {
        int kk = tid / 3, c = tid - kk * 3;
        int p = idx[((size_t)b * NPTS + n) * KNN_K + kk];
        nbr[kk][c] = xb[c * NPTS + p];
    }
    __syncthreads();
    int o = tid;
    float wv[6];
    #pragma unroll
    for (int c = 0; c < 6; ++c) wv[c] = W1[o * 6 + c];
    float mx = -3.4e38f, ls = 0.f, lq = 0.f;
    for (int kk = 0; kk < KNN_K; ++kk) {
        float h = 0.f;
        #pragma unroll
        for (int c = 0; c < 3; ++c) h = fmaf(wv[c], nbr[kk][c] - ctr[c], h);
        #pragma unroll
        for (int c = 0; c < 3; ++c) h = fmaf(wv[3 + c], ctr[c], h);
        mx = fmaxf(mx, h); ls += h; lq = fmaf(h, h, lq);
    }
    hmaxEnc[((size_t)b * 64 + o) * NPTS + n] = encodeF(mx);
    atomicAdd(&chanSum[o], ls);
    atomicAdd(&chanSumSq[o], lq);
}

// ---------- WMMA edge GEMM: gather->LDS fp16, packed-weight v16h A loads,
//            fused k-max (global_atomic_max_u32) + BN stats (shfl + ds_add) ----------
template <int C2, int O>
__global__ __launch_bounds__(128) void edge_gemm_kernel(const float* __restrict__ x,
                                                        const int* __restrict__ idx,
                                                        const v16h* __restrict__ Wfrag,
                                                        unsigned* __restrict__ hmaxEnc,
                                                        float* __restrict__ chanSum,
                                                        float* __restrict__ chanSumSq) {
    constexpr int C  = C2 / 2;
    constexpr int LD = C2 + 8;                  // pad (keeps rows 16B aligned, C2 mult of 32)
    constexpr int KT = C2 / 32;
    __shared__ __align__(16) _Float16 fTile[16][LD];   // [col][channel]
    __shared__ float redS[O], redQ[O];
    int b = blockIdx.y;
    int colBase = blockIdx.x * 16;              // columns are (n,kk) pairs
    const float* xb = x + (size_t)b * C * NPTS;

    for (int t = threadIdx.x; t < O; t += 128) { redS[t] = 0.f; redQ[t] = 0.f; }

    // gather graph feature tile, two channels per thread -> one b32 LDS store
    for (int t = threadIdx.x; t < 16 * C; t += 128) {
        int e = t & 15;
        int c = (t >> 4) * 2;
        int gcol = colBase + e;
        int n = gcol / KNN_K, kk = gcol - n * KNN_K;
        float v0, v1;
        if (c < C) {
            int p = idx[((size_t)b * NPTS + n) * KNN_K + kk];
            v0 = xb[c * NPTS + p] - xb[c * NPTS + n];
            v1 = xb[(c + 1) * NPTS + p] - xb[(c + 1) * NPTS + n];
        } else {
            v0 = xb[(c - C) * NPTS + n];
            v1 = xb[(c - C + 1) * NPTS + n];
        }
        union { _Float16 h[2]; unsigned u; } pk;
        pk.h[0] = (_Float16)v0; pk.h[1] = (_Float16)v1;
        *(unsigned*)&fTile[e][c] = pk.u;
    }
    __syncthreads();

    int wave = threadIdx.x >> 5, lane = threadIdx.x & 31;
    int row = lane & 15;          // A-row / B-col / D-col per ISA layouts
    int hi  = lane >> 4;          // lane-half selects K-halves / M+8

    for (int mt = wave; mt < O / 16; mt += 4) {
        v8f acc = {};
        const v16h* wp = Wfrag + (size_t)mt * KT * 32 + lane;
        __builtin_prefetch((const void*)(wp + 4 * (size_t)KT * 32), 0, 0); // next mt for this wave
        #pragma unroll
        for (int kt = 0; kt < KT; ++kt) {
            v16h a  = wp[kt * 32];                                        // 32B global load
            v16h bf = *(const v16h_a16*)&fTile[row][kt * 32 + hi * 16];   // 2x ds_load_b128
            acc = __builtin_amdgcn_wmma_f32_16x16x32_f16(false, a, false, bf,
                                                         (short)0, acc, false, false);
        }
        // D layout: VGPR r -> M = r + 8*hi, col = lane&15
        int n = (colBase + row) / KNN_K;
        #pragma unroll
        for (int r = 0; r < 8; ++r) {
            float h = acc[r];
            int o = mt * 16 + r + hi * 8;
            atomicMax(&hmaxEnc[((size_t)b * O + o) * NPTS + n], encodeF(h)); // max over k
            float s = h, q = h * h;                 // reduce across 16 col-lanes first
            #pragma unroll
            for (int off = 1; off < 16; off <<= 1) {
                s += __shfl_xor(s, off, 32);
                q += __shfl_xor(q, off, 32);
            }
            if (row == 0) { atomicAdd(&redS[o], s); atomicAdd(&redQ[o], q); }
        }
    }
    __syncthreads();
    for (int t = threadIdx.x; t < O; t += 128) {
        atomicAdd(&chanSum[t], redS[t]);
        atomicAdd(&chanSumSq[t], redQ[t]);
    }
}

// ---------- BN finalize + LeakyReLU on the k-maxed tensor ----------
__global__ void edge_bn_kernel(const unsigned* __restrict__ hmaxEnc,
                               const float* __restrict__ chanSum,
                               const float* __restrict__ chanSumSq,
                               const float* __restrict__ g, const float* __restrict__ bta,
                               float* __restrict__ xout, int O, float invCount) {
    int i = blockIdx.x * blockDim.x + threadIdx.x;
    if (i >= BATCH * O * NPTS) return;
    int o = (i / NPTS) % O;
    float m   = chanSum[o] * invCount;
    float var = chanSumSq[o] * invCount - m * m;
    float s   = g[o] * __frsqrt_rn(var + BNEPS);
    float h   = decodeF(hmaxEnc[i]);
    xout[i] = lrelu((h - m) * s + bta[o]);
}

// ---------- W5 conv (1024x512 GEMM over concat features) with WMMA ----------
__global__ __launch_bounds__(128) void conv5_kernel(const float* __restrict__ x1,
                                                    const float* __restrict__ x2,
                                                    const float* __restrict__ x3,
                                                    const float* __restrict__ x4,
                                                    const v16h* __restrict__ Wfrag,
                                                    _Float16* __restrict__ h5,
                                                    float* __restrict__ chanSum,
                                                    float* __restrict__ chanSumSq) {
    constexpr int KT = 16;                       // 512/32
    __shared__ __align__(16) _Float16 fTile[16][512 + 8];
    __shared__ float redS[1024], redQ[1024];
    int b = blockIdx.y, colBase = blockIdx.x * 16;

    for (int t = threadIdx.x; t < 1024; t += 128) { redS[t] = 0.f; redQ[t] = 0.f; }

    for (int t = threadIdx.x; t < 16 * 256; t += 128) {  // two channels per thread
        int e = t & 15;
        int c = (t >> 4) * 2;
        int n = colBase + e;
        const float* src;
        if (c < 64)       src = &x1[((size_t)b * 64  +  c       ) * NPTS + n];
        else if (c < 128) src = &x2[((size_t)b * 64  + (c - 64 )) * NPTS + n];
        else if (c < 256) src = &x3[((size_t)b * 128 + (c - 128)) * NPTS + n];
        else              src = &x4[((size_t)b * 256 + (c - 256)) * NPTS + n];
        union { _Float16 h[2]; unsigned u; } pk;
        pk.h[0] = (_Float16)src[0];
        pk.h[1] = (_Float16)src[NPTS];
        *(unsigned*)&fTile[e][c] = pk.u;
    }
    __syncthreads();

    int wave = threadIdx.x >> 5, lane = threadIdx.x & 31;
    int row = lane & 15, hi = lane >> 4;

    for (int mt = wave; mt < 64; mt += 4) {
        v8f acc = {};
        const v16h* wp = Wfrag + (size_t)mt * KT * 32 + lane;
        __builtin_prefetch((const void*)(wp + 4 * (size_t)KT * 32), 0, 0);
        #pragma unroll
        for (int kt = 0; kt < KT; ++kt) {
            v16h a  = wp[kt * 32];
            v16h bf = *(const v16h_a16*)&fTile[row][kt * 32 + hi * 16];
            acc = __builtin_amdgcn_wmma_f32_16x16x32_f16(false, a, false, bf,
                                                         (short)0, acc, false, false);
        }
        int n = colBase + row;
        #pragma unroll
        for (int r = 0; r < 8; ++r) {
            float h = acc[r];
            int o = mt * 16 + r + hi * 8;
            h5[((size_t)b * 1024 + o) * NPTS + n] = (_Float16)h;
            float s = h, q = h * h;
            #pragma unroll
            for (int off = 1; off < 16; off <<= 1) {
                s += __shfl_xor(s, off, 32);
                q += __shfl_xor(q, off, 32);
            }
            if (row == 0) { atomicAdd(&redS[o], s); atomicAdd(&redQ[o], q); }
        }
    }
    __syncthreads();
    for (int t = threadIdx.x; t < 1024; t += 128) {
        atomicAdd(&chanSum[t], redS[t]);
        atomicAdd(&chanSumSq[t], redQ[t]);
    }
}

// ---------- BN + lrelu + global max/mean pool over N ----------
__global__ __launch_bounds__(256) void pool_kernel(const _Float16* __restrict__ h5,
                                                   const float* __restrict__ chanSum,
                                                   const float* __restrict__ chanSumSq,
                                                   const float* __restrict__ g,
                                                   const float* __restrict__ bta,
                                                   float* __restrict__ pooled) {
    __shared__ float smx[256], ssm[256];
    int o = blockIdx.x, b = blockIdx.y, tid = threadIdx.x;
    float invC = 1.f / (float)(BATCH * NPTS);
    float m   = chanSum[o] * invC;
    float var = chanSumSq[o] * invC - m * m;
    float s   = g[o] * __frsqrt_rn(var + BNEPS);
    float bb  = bta[o];
    float mx = -3.4e38f, sm = 0.f;
    const _Float16* hp = h5 + ((size_t)b * 1024 + o) * NPTS;
    for (int n = tid; n < NPTS; n += 256) {
        float y = lrelu(((float)hp[n] - m) * s + bb);
        mx = fmaxf(mx, y); sm += y;
    }
    smx[tid] = mx; ssm[tid] = sm; __syncthreads();
    for (int off = 128; off > 0; off >>= 1) {
        if (tid < off) { smx[tid] = fmaxf(smx[tid], smx[tid + off]); ssm[tid] += ssm[tid + off]; }
        __syncthreads();
    }
    if (tid == 0) {
        pooled[(size_t)b * 2048 + o]        = smx[0];
        pooled[(size_t)b * 2048 + 1024 + o] = ssm[0] * (1.f / NPTS);
    }
}

// ---------- tiny FC GEMV (8 rows) ----------
__global__ void fc_kernel(const float* __restrict__ in, const float* __restrict__ W,
                          const float* __restrict__ bias, float* __restrict__ out,
                          int On, int Cn) {
    int i = blockIdx.x * blockDim.x + threadIdx.x;
    if (i >= BATCH * On) return;
    int b = i / On, o = i - b * On;
    float s = bias ? bias[o] : 0.f;
    const float* ip = in + (size_t)b * Cn;
    const float* wp = W  + (size_t)o * Cn;
    for (int c = 0; c < Cn; ++c) s = fmaf(ip[c], wp[c], s);
    out[i] = s;
}

// ---------- batch-stat BN over B=8 rows + lrelu ----------
__global__ void bnfc_kernel(const float* __restrict__ z, const float* __restrict__ g,
                            const float* __restrict__ bta, float* __restrict__ out, int On) {
    int o = blockIdx.x * blockDim.x + threadIdx.x;
    if (o >= On) return;
    float s = 0.f, q = 0.f;
    for (int b = 0; b < BATCH; ++b) { float v = z[b * On + o]; s += v; q = fmaf(v, v, q); }
    float m = s * (1.f / BATCH);
    float var = q * (1.f / BATCH) - m * m;
    float sc = g[o] * __frsqrt_rn(var + BNEPS);
    for (int b = 0; b < BATCH; ++b) out[b * On + o] = lrelu((z[b * On + o] - m) * sc + bta[o]);
}

// ==================== host side ====================
extern "C" void kernel_launch(void* const* d_in, const int* in_sizes, int n_in,
                              void* d_out, int out_size, void* d_ws, size_t ws_size,
                              hipStream_t stream) {
    (void)in_sizes; (void)n_in; (void)out_size; (void)ws_size;

    const float* x   = (const float*)d_in[0];
    const float* W1  = (const float*)d_in[1];
    const float* g1  = (const float*)d_in[2];
    const float* b1  = (const float*)d_in[3];
    const float* W2  = (const float*)d_in[4];
    const float* g2  = (const float*)d_in[5];
    const float* b2  = (const float*)d_in[6];
    const float* W3  = (const float*)d_in[7];
    const float* g3  = (const float*)d_in[8];
    const float* b3  = (const float*)d_in[9];
    const float* W4  = (const float*)d_in[10];
    const float* g4  = (const float*)d_in[11];
    const float* b4  = (const float*)d_in[12];
    const float* W5  = (const float*)d_in[13];
    const float* g5  = (const float*)d_in[14];
    const float* b5  = (const float*)d_in[15];
    const float* Wl1 = (const float*)d_in[16];
    const float* g6  = (const float*)d_in[17];
    const float* b6  = (const float*)d_in[18];
    const float* Wl2 = (const float*)d_in[19];
    const float* bl2 = (const float*)d_in[20];
    const float* g7  = (const float*)d_in[21];
    const float* b7  = (const float*)d_in[22];
    const float* Wl3 = (const float*)d_in[23];
    const float* bl3 = (const float*)d_in[24];

    char* wsb = (char*)d_ws;
    size_t off = 0;
    auto alloc = [&](size_t bytes) -> void* {
        void* p = wsb + off;
        off += (bytes + 255) & ~(size_t)255;
        return p;
    };
    int*      idx    = (int*)     alloc((size_t)BATCH * NPTS * KNN_K * sizeof(int));
    float*    xx     = (float*)   alloc((size_t)BATCH * NPTS * sizeof(float));
    unsigned* hmax   = (unsigned*)alloc((size_t)BATCH * 256 * NPTS * sizeof(unsigned));
    float*    cSum   = (float*)   alloc(1024 * sizeof(float));
    float*    cSq    = (float*)   alloc(1024 * sizeof(float));
    float*    x1     = (float*)   alloc((size_t)BATCH * 64  * NPTS * sizeof(float));
    float*    x2b    = (float*)   alloc((size_t)BATCH * 64  * NPTS * sizeof(float));
    float*    x3b    = (float*)   alloc((size_t)BATCH * 128 * NPTS * sizeof(float));
    float*    x4b    = (float*)   alloc((size_t)BATCH * 256 * NPTS * sizeof(float));
    _Float16* h5     = (_Float16*)alloc((size_t)BATCH * 1024 * NPTS * sizeof(_Float16));
    float*    pooled = (float*)   alloc((size_t)BATCH * 2048 * sizeof(float));
    float*    z1     = (float*)   alloc((size_t)BATCH * 512 * sizeof(float));
    float*    a1     = (float*)   alloc((size_t)BATCH * 512 * sizeof(float));
    float*    z2     = (float*)   alloc((size_t)BATCH * 256 * sizeof(float));
    float*    a2     = (float*)   alloc((size_t)BATCH * 256 * sizeof(float));
    _Float16* Wp2    = (_Float16*)alloc((size_t)64   * 128 * sizeof(_Float16));
    _Float16* Wp3    = (_Float16*)alloc((size_t)128  * 128 * sizeof(_Float16));
    _Float16* Wp4    = (_Float16*)alloc((size_t)256  * 256 * sizeof(_Float16));
    _Float16* Wp5    = (_Float16*)alloc((size_t)1024 * 512 * sizeof(_Float16));

    const int   colTiles = NPTS * KNN_K / 16;      // 2560
    const float invEdge  = 1.f / (float)(BATCH * NPTS * KNN_K);

    // ---- pack all WMMA weights once (fp32 -> fragment-ordered fp16) ----
    packW_kernel<<<(64 * 128 + 255) / 256, 256, 0, stream>>>(W2, Wp2, 64, 128);
    packW_kernel<<<(128 * 128 + 255) / 256, 256, 0, stream>>>(W3, Wp3, 128, 128);
    packW_kernel<<<(256 * 256 + 255) / 256, 256, 0, stream>>>(W4, Wp4, 256, 256);
    packW_kernel<<<(1024 * 512 + 255) / 256, 256, 0, stream>>>(W5, Wp5, 1024, 512);

    // ---- layer 1 (C=3 -> 64) ----
    hipMemsetAsync(cSum, 0, 1024 * 4, stream);
    hipMemsetAsync(cSq,  0, 1024 * 4, stream);
    sumsq_kernel<<<(BATCH * NPTS + 255) / 256, 256, 0, stream>>>(x, xx, 3);
    knn_kernel<<<dim3(NPTS / 4, BATCH), 128, 0, stream>>>(x, xx, idx, 3);
    edge1_kernel<<<dim3(NPTS, BATCH), 64, 0, stream>>>(x, idx, W1, hmax, cSum, cSq);
    edge_bn_kernel<<<(BATCH * 64 * NPTS + 255) / 256, 256, 0, stream>>>(hmax, cSum, cSq, g1, b1, x1, 64, invEdge);

    // ---- layer 2 (C=64 -> 64) ----
    hipMemsetAsync(cSum, 0, 1024 * 4, stream);
    hipMemsetAsync(cSq,  0, 1024 * 4, stream);
    hipMemsetAsync(hmax, 0, (size_t)BATCH * 64 * NPTS * 4, stream);
    sumsq_kernel<<<(BATCH * NPTS + 255) / 256, 256, 0, stream>>>(x1, xx, 64);
    knn_kernel<<<dim3(NPTS / 4, BATCH), 128, 0, stream>>>(x1, xx, idx, 64);
    edge_gemm_kernel<128, 64><<<dim3(colTiles, BATCH), 128, 0, stream>>>(x1, idx, (const v16h*)Wp2, hmax, cSum, cSq);
    edge_bn_kernel<<<(BATCH * 64 * NPTS + 255) / 256, 256, 0, stream>>>(hmax, cSum, cSq, g2, b2, x2b, 64, invEdge);

    // ---- layer 3 (C=64 -> 128) ----
    hipMemsetAsync(cSum, 0, 1024 * 4, stream);
    hipMemsetAsync(cSq,  0, 1024 * 4, stream);
    hipMemsetAsync(hmax, 0, (size_t)BATCH * 128 * NPTS * 4, stream);
    sumsq_kernel<<<(BATCH * NPTS + 255) / 256, 256, 0, stream>>>(x2b, xx, 64);
    knn_kernel<<<dim3(NPTS / 4, BATCH), 128, 0, stream>>>(x2b, xx, idx, 64);
    edge_gemm_kernel<128, 128><<<dim3(colTiles, BATCH), 128, 0, stream>>>(x2b, idx, (const v16h*)Wp3, hmax, cSum, cSq);
    edge_bn_kernel<<<(BATCH * 128 * NPTS + 255) / 256, 256, 0, stream>>>(hmax, cSum, cSq, g3, b3, x3b, 128, invEdge);

    // ---- layer 4 (C=128 -> 256) ----
    hipMemsetAsync(cSum, 0, 1024 * 4, stream);
    hipMemsetAsync(cSq,  0, 1024 * 4, stream);
    hipMemsetAsync(hmax, 0, (size_t)BATCH * 256 * NPTS * 4, stream);
    sumsq_kernel<<<(BATCH * NPTS + 255) / 256, 256, 0, stream>>>(x3b, xx, 128);
    knn_kernel<<<dim3(NPTS / 4, BATCH), 128, 0, stream>>>(x3b, xx, idx, 128);
    edge_gemm_kernel<256, 256><<<dim3(colTiles, BATCH), 128, 0, stream>>>(x3b, idx, (const v16h*)Wp4, hmax, cSum, cSq);
    edge_bn_kernel<<<(BATCH * 256 * NPTS + 255) / 256, 256, 0, stream>>>(hmax, cSum, cSq, g4, b4, x4b, 256, invEdge);

    // ---- head: W5 conv + BN + lrelu + max/mean pool ----
    hipMemsetAsync(cSum, 0, 1024 * 4, stream);
    hipMemsetAsync(cSq,  0, 1024 * 4, stream);
    conv5_kernel<<<dim3(NPTS / 16, BATCH), 128, 0, stream>>>(x1, x2b, x3b, x4b, (const v16h*)Wp5, h5, cSum, cSq);
    pool_kernel<<<dim3(1024, BATCH), 256, 0, stream>>>(h5, cSum, cSq, g5, b5, pooled);

    // ---- FC stack ----
    fc_kernel<<<(BATCH * 512 + 255) / 256, 256, 0, stream>>>(pooled, Wl1, nullptr, z1, 512, 2048);
    bnfc_kernel<<<2, 256, 0, stream>>>(z1, g6, b6, a1, 512);
    fc_kernel<<<(BATCH * 256 + 255) / 256, 256, 0, stream>>>(a1, Wl2, bl2, z2, 256, 512);
    bnfc_kernel<<<1, 256, 0, stream>>>(z2, g7, b7, a2, 256);
    fc_kernel<<<(BATCH * 40 + 255) / 256, 256, 0, stream>>>(a2, Wl3, bl3, (float*)d_out, 40, 256);
}
